// GNN_norm_65807488909361
// MI455X (gfx1250) — compile-verified, matrix operand order checked by reading, not run on hardware
//
#include <hip/hip_runtime.h>
#include <math.h>

typedef __attribute__((ext_vector_type(2)))  float    v2f;
typedef __attribute__((ext_vector_type(8)))  float    v8f;
typedef __attribute__((ext_vector_type(16))) _Float16 v16h;

#ifndef __has_builtin
#define __has_builtin(x) 0
#endif

// ---------------------------------------------------------------------------
// Degree / normalization
// ---------------------------------------------------------------------------
__global__ void k_deg_init(float* __restrict__ deg, int n) {
  int i = blockIdx.x * blockDim.x + threadIdx.x;
  if (i < n) deg[i] = 1.0f;                      // self-loop weight 1
}

__global__ void k_deg_scatter(const int* __restrict__ dst,
                              const float* __restrict__ w,
                              float* __restrict__ deg, int E) {
  int e = blockIdx.x * blockDim.x + threadIdx.x;
  if (e < E) atomicAdd(&deg[dst[e]], w[e]);
}

__global__ void k_dinv(const float* __restrict__ deg,
                       float* __restrict__ dinv, int n) {
  int i = blockIdx.x * blockDim.x + threadIdx.x;
  if (i < n) {
    float d = deg[i];
    dinv[i] = d > 0.0f ? rsqrtf(d) : 0.0f;
  }
}

// ---------------------------------------------------------------------------
// WMMA GEMM: Y[n,16] = X[n,K] @ W[16,K]^T   (one 16-row tile per wave32)
// V_WMMA_F32_16X16X4_F32 (full fp32 precision).
// ---------------------------------------------------------------------------
__global__ void k_gemm16(const float* __restrict__ X,
                         const float* __restrict__ W,
                         float* __restrict__ Y, int n, int K) {
  int lane = threadIdx.x & 31;
  int wave = threadIdx.x >> 5;
  int tile = blockIdx.x * (blockDim.x >> 5) + wave;
  int ntiles = (n + 15) >> 4;
  if (tile >= ntiles) return;                    // wave-uniform exit: EXEC stays full
  int m0 = tile << 4;
  int m  = lane & 15;                            // row within tile (A) / col (B,D)
  int row = m0 + m; if (row >= n) row = n - 1;   // clamp loads (tail tile safety)
  v8f acc = {};

#if __has_builtin(__builtin_amdgcn_wmma_f32_16x16x4_f32)
  // A 16x4 f32 layout: lanes 0-15 hold K={0,1}, lanes 16-31 K={2,3}
  int kb = (lane >> 4) << 1;                     // 0 or 2
  const float* xrow = X + (size_t)row * K;
  const float* wrow = W + (size_t)m * K;         // B[k][n] = W[n*K+k]; n == lane&15
  for (int k0 = 0; k0 < K; k0 += 4) {
    v2f a, b;
    a.x = xrow[k0 + kb];
    a.y = xrow[k0 + kb + 1];
    b.x = wrow[k0 + kb];
    b.y = wrow[k0 + kb + 1];
    acc = __builtin_amdgcn_wmma_f32_16x16x4_f32(
        false, a, false, b, (short)0, acc, false, false);
  }
#else
  // Fallback: f16 WMMA 16x16x32 (A/B converted, f32 accumulate)
  const float* xrow = X + (size_t)row * K;
  const float* wrow = W + (size_t)m * K;
  int hi = (lane >> 4) << 3;                     // +8 K-offset for lanes 16-31
  for (int k0 = 0; k0 < K; k0 += 32) {
    v16h a, b;
#pragma unroll
    for (int v = 0; v < 8; ++v) {
      int kp = (v < 4 ? 2 * v : 2 * (v - 4) + 16) + hi + k0;
      a[2 * v]     = (kp     < K) ? (_Float16)xrow[kp]     : (_Float16)0.0f;
      a[2 * v + 1] = (kp + 1 < K) ? (_Float16)xrow[kp + 1] : (_Float16)0.0f;
      b[2 * v]     = (kp     < K) ? (_Float16)wrow[kp]     : (_Float16)0.0f;
      b[2 * v + 1] = (kp + 1 < K) ? (_Float16)wrow[kp + 1] : (_Float16)0.0f;
    }
    acc = __builtin_amdgcn_wmma_f32_16x16x32_f16(
        false, a, false, b, (short)0, acc, false, false);
  }
#endif

  // D layout: VGPR r, lanes 0-15 -> M=r, lanes 16-31 -> M=r+8; N = lane&15
  int mhi = (lane >> 4) << 3;
  int nn  = lane & 15;
  float* yout = Y + (size_t)m0 * 16;
#pragma unroll
  for (int r = 0; r < 8; ++r) {
    int mr = r + mhi;
    if (m0 + mr < n) yout[(size_t)mr * 16 + nn] = acc[r];
  }
}

// ---------------------------------------------------------------------------
// Aggregation (16 features): self-loop seed, edge scatter (16 lanes/edge),
// bias(+ReLU)
// ---------------------------------------------------------------------------
__global__ void k_agg_init16(const float* __restrict__ lin,
                             const float* __restrict__ dinv,
                             float* __restrict__ agg, int n) {
  int i = blockIdx.x * blockDim.x + threadIdx.x;
  int node = i >> 4;
  if (node < n) {
    float s = dinv[node];
    agg[i] = s * s * lin[i];
  }
}

// 16 threads per edge: lane-per-feature => coalesced gather of lin[src] and
// contiguous (mergeable) atomic adds into agg[dst].
__global__ void k_agg_edge16(const int* __restrict__ src,
                             const int* __restrict__ dst,
                             const float* __restrict__ w,
                             const float* __restrict__ dinv,
                             const float* __restrict__ lin,
                             float* __restrict__ agg, int E) {
  int t = blockIdx.x * blockDim.x + threadIdx.x;
  int e = t >> 4;
  if (e >= E) return;
  int f = t & 15;
  int s = src[e], d = dst[e];                 // uniform per half-wave -> broadcast
  float c = dinv[s] * w[e] * dinv[d];
  atomicAdd(&agg[(size_t)d * 16 + f], c * lin[(size_t)s * 16 + f]);
}

__global__ void k_bias_relu16(float* __restrict__ agg,
                              const float* __restrict__ b, int n, int relu) {
  int i = blockIdx.x * blockDim.x + threadIdx.x;
  int node = i >> 4;
  if (node < n) {
    float v = agg[i] + b[i & 15];
    agg[i] = relu ? fmaxf(v, 0.0f) : v;
  }
}

// ---------------------------------------------------------------------------
// Fused: layer-2 bias+ReLU  ->  layer-3 linear (dot W3)  ->  self-loop seed.
// In-place updates AGG to h2; writes clin and cagg seed in the same pass.
// ---------------------------------------------------------------------------
__global__ void k_finish2_dot(float* __restrict__ agg,
                              const float* __restrict__ b2,
                              const float* __restrict__ W3,
                              const float* __restrict__ dinv,
                              float* __restrict__ clin,
                              float* __restrict__ cagg, int n) {
  int i = blockIdx.x * blockDim.x + threadIdx.x;
  if (i >= n) return;
  float* row = agg + (size_t)i * 16;
  float acc = 0.0f;
#pragma unroll
  for (int f = 0; f < 16; ++f) {
    float v = fmaxf(row[f] + b2[f], 0.0f);
    row[f] = v;
    acc += v * W3[f];
  }
  clin[i] = acc;
  float s = dinv[i];
  cagg[i] = s * s * acc;
}

__global__ void k_agg_edge1(const int* __restrict__ src,
                            const int* __restrict__ dst,
                            const float* __restrict__ w,
                            const float* __restrict__ dinv,
                            const float* __restrict__ lin,
                            float* __restrict__ agg, int E) {
  int e = blockIdx.x * blockDim.x + threadIdx.x;
  if (e >= E) return;
  int s = src[e], d = dst[e];
  float c = dinv[s] * w[e] * dinv[d];
  atomicAdd(&agg[d], c * lin[s]);
}

__global__ void k_bias1(float* __restrict__ agg,
                        const float* __restrict__ b, int n) {
  int i = blockIdx.x * blockDim.x + threadIdx.x;
  if (i < n) agg[i] += b[0];
}

// ---------------------------------------------------------------------------
// Softmax over n logits (single-block reductions)
// ---------------------------------------------------------------------------
__global__ void k_max(const float* __restrict__ c, float* __restrict__ out, int n) {
  __shared__ float sm[1024];
  float m = -3.402823466e38f;
  for (int i = threadIdx.x; i < n; i += blockDim.x) m = fmaxf(m, c[i]);
  sm[threadIdx.x] = m;
  __syncthreads();
  for (int s = blockDim.x >> 1; s > 0; s >>= 1) {
    if ((int)threadIdx.x < s) sm[threadIdx.x] = fmaxf(sm[threadIdx.x], sm[threadIdx.x + s]);
    __syncthreads();
  }
  if (threadIdx.x == 0) out[0] = sm[0];
}

__global__ void k_sumexp(const float* __restrict__ c, const float* __restrict__ mx,
                         float* __restrict__ out, int n) {
  __shared__ float sm[1024];
  float m = mx[0], acc = 0.0f;
  for (int i = threadIdx.x; i < n; i += blockDim.x) acc += expf(c[i] - m);
  sm[threadIdx.x] = acc;
  __syncthreads();
  for (int s = blockDim.x >> 1; s > 0; s >>= 1) {
    if ((int)threadIdx.x < s) sm[threadIdx.x] += sm[threadIdx.x + s];
    __syncthreads();
  }
  if (threadIdx.x == 0) out[0] = sm[0];
}

__global__ void k_softmax(const float* __restrict__ c, const float* __restrict__ mx,
                          const float* __restrict__ sum, float* __restrict__ choice, int n) {
  int i = blockIdx.x * blockDim.x + threadIdx.x;
  if (i < n) choice[i] = expf(c[i] - mx[0]) / sum[0];
}

// ---------------------------------------------------------------------------
// value = mean(h2, axis=0) @ A2w^T + A2b     (single block)
// ---------------------------------------------------------------------------
__global__ void k_value(const float* __restrict__ h2, const float* __restrict__ A2w,
                        const float* __restrict__ A2b, float* __restrict__ out, int n) {
  __shared__ float sm[256];
  int f = threadIdx.x & 15, part = threadIdx.x >> 4;   // 16 partials per feature
  float acc = 0.0f;
  for (int i = part; i < n; i += 16) acc += h2[(size_t)i * 16 + f];
  sm[threadIdx.x] = acc;
  __syncthreads();
  for (int s = 128; s >= 16; s >>= 1) {
    if ((int)threadIdx.x < s) sm[threadIdx.x] += sm[threadIdx.x + s];
    __syncthreads();
  }
  if (threadIdx.x < 16) sm[threadIdx.x] = (sm[threadIdx.x] / (float)n) * A2w[threadIdx.x];
  __syncthreads();
  if (threadIdx.x == 0) {
    float v = A2b[0];
#pragma unroll
    for (int f2 = 0; f2 < 16; ++f2) v += sm[f2];
    out[0] = v;
  }
}

// ---------------------------------------------------------------------------
// Host launcher
// ---------------------------------------------------------------------------
static inline int cdiv(int a, int b) { return (a + b - 1) / b; }

extern "C" void kernel_launch(void* const* d_in, const int* in_sizes, int n_in,
                              void* d_out, int out_size, void* d_ws, size_t ws_size,
                              hipStream_t stream) {
  const float* x    = (const float*)d_in[0];
  const int*   edge = (const int*)  d_in[1];
  const float* w    = (const float*)d_in[2];
  const float* W1   = (const float*)d_in[3];
  const float* b1   = (const float*)d_in[4];
  const float* W2   = (const float*)d_in[5];
  const float* b2   = (const float*)d_in[6];
  const float* W3   = (const float*)d_in[7];
  const float* b3   = (const float*)d_in[8];
  const float* A2w  = (const float*)d_in[9];
  const float* A2b  = (const float*)d_in[10];

  const int N = in_sizes[0] / 64;
  const int E = in_sizes[2];
  const int* src = edge;
  const int* dst = edge + E;

  float* ws   = (float*)d_ws;
  float* deg  = ws;
  float* dinv = ws + (size_t)N;
  float* LIN  = ws + (size_t)2  * N;   // 16*N
  float* AGG  = ws + (size_t)18 * N;   // 16*N
  float* CLIN = ws + (size_t)34 * N;
  float* CAGG = ws + (size_t)35 * N;
  float* MX   = ws + (size_t)36 * N;
  float* SUM  = MX + 1;

  float* choice = (float*)d_out;
  float* value  = choice + N;

  const int T = 256;
  const int wavesPerBlk = T / 32;
  const int ntiles = cdiv(N, 16);
  const int gemmBlocks = cdiv(ntiles, wavesPerBlk);

  // --- degree normalization (shared by all layers) ---
  k_deg_init   <<<cdiv(N, T), T, 0, stream>>>(deg, N);
  k_deg_scatter<<<cdiv(E, T), T, 0, stream>>>(dst, w, deg, E);
  k_dinv       <<<cdiv(N, T), T, 0, stream>>>(deg, dinv, N);

  // --- layer 1: lin = x @ W1^T; agg; +b1; ReLU ---
  k_gemm16     <<<gemmBlocks, T, 0, stream>>>(x, W1, LIN, N, 64);
  k_agg_init16 <<<cdiv(16 * N, T), T, 0, stream>>>(LIN, dinv, AGG, N);
  k_agg_edge16 <<<cdiv(16 * E, T), T, 0, stream>>>(src, dst, w, dinv, LIN, AGG, E);
  k_bias_relu16<<<cdiv(16 * N, T), T, 0, stream>>>(AGG, b1, N, 1);

  // --- layer 2: lin = h1 @ W2^T; agg; +b2; ReLU (fused below) ---
  k_gemm16     <<<gemmBlocks, T, 0, stream>>>(AGG, W2, LIN, N, 16);
  k_agg_init16 <<<cdiv(16 * N, T), T, 0, stream>>>(LIN, dinv, AGG, N);
  k_agg_edge16 <<<cdiv(16 * E, T), T, 0, stream>>>(src, dst, w, dinv, LIN, AGG, E);

  // --- fused: +b2,ReLU -> h2; clin = h2 @ W3^T; cagg seed ---
  k_finish2_dot<<<cdiv(N, T), T, 0, stream>>>(AGG, b2, W3, dinv, CLIN, CAGG, N);

  // --- layer 3 edge aggregation (1 feature) + b3 ---
  k_agg_edge1  <<<cdiv(E, T), T, 0, stream>>>(src, dst, w, dinv, CLIN, CAGG, E);
  k_bias1      <<<cdiv(N, T), T, 0, stream>>>(CAGG, b3, N);

  // --- softmax over nodes -> choice ---
  k_max    <<<1, 1024, 0, stream>>>(CAGG, MX, N);
  k_sumexp <<<1, 1024, 0, stream>>>(CAGG, MX, SUM, N);
  k_softmax<<<cdiv(N, T), T, 0, stream>>>(CAGG, MX, SUM, choice, N);

  // --- value head: mean-pool h2, linear ---
  k_value<<<1, 256, 0, stream>>>(AGG, A2w, A2b, value, N);
}
